// InputBlock_26938034880915
// MI455X (gfx1250) — matrix operand order repaired and checked by reading, not compiled, not run on hardware
//
#include <hip/hip_runtime.h>
#include <hip/hip_bf16.h>
#include <math.h>

typedef __attribute__((ext_vector_type(16))) __bf16 v16bf;
typedef __attribute__((ext_vector_type(8)))  __bf16 v8bf;
typedef __attribute__((ext_vector_type(4)))  __bf16 v4bf;
typedef __attribute__((ext_vector_type(8)))  float  v8f;
typedef __attribute__((ext_vector_type(4)))  float  v4f;

#define E_DIM 256
#define H_DIM 128
#define K_NB  32
#define TOK_PER_BLK 128
#define NODES_PER_BLK 4
#define EPSV 1e-5f

// shared-memory layout (bytes), all 16B aligned
#define OFF_WLINT 0                       // bf16 [128][256]  W_lin^T
#define OFF_W1T   65536                   // bf16 [128][128]  W1^T
#define OFF_W2T   98304                   // bf16 [128][128]  W2^T
#define OFF_SA    131072                  // bf16 [128][256]  A tile (edge feats)
#define OFF_SHID  196608                  // f32  [128][132]  edge_hidden / edge_out
#define OFF_SX    264192                  // bf16 [128][136]  LN(h) then gelu(xW1+b1)
#define OFF_SNODE 299008                  // f32  [4][128]    node sums
#define OFF_BIAS  301056                  // f32  7*[128]     biases/gains
#define OFF_NSTAT 304640                  // f32  mu[4], rs[4]
#define OFF_MASK  304704                  // int  [128]
#define SMEM_BYTES 305216

#define SHID_STRIDE 132
#define SX_STRIDE   136

// A fragment (16x32 bf16, M rows): lane L -> row = row0+(L&15),
// K chunks [c..c+7] and [c+16..c+23] with c = k0 + (L>=16 ? 8 : 0)
__device__ __forceinline__ v16bf load_a_frag(const __bf16* base, int stride,
                                             int row0, int k0, int lane) {
    const int m = lane & 15;
    const int c = k0 + ((lane >> 4) << 3);
    const __bf16* p = base + (row0 + m) * stride + c;
    union { v16bf v; v8bf h[2]; } u;
    u.h[0] = *(const v8bf*)(p);
    u.h[1] = *(const v8bf*)(p + 16);
    return u.v;
}

// B fragment (32x16 bf16, from B^T stored [n][k]): lane L -> col = n0+(L&15),
// K chunks [c..c+7] and [c+16..c+23] with c = k0 + (L>=16 ? 8 : 0)
__device__ __forceinline__ v16bf load_b_frag(const __bf16* baseT, int stride,
                                             int n0, int k0, int lane) {
    const int n = n0 + (lane & 15);
    const int c = k0 + ((lane >> 4) << 3);
    const __bf16* p = baseT + n * stride + c;
    union { v16bf v; v8bf h[2]; } u;
    u.h[0] = *(const v8bf*)(p);
    u.h[1] = *(const v8bf*)(p + 16);
    return u.v;
}

// one wave: 16-row M-tile x all 8 N-tiles, K = KSTEPS*32.
// Software-pipelined: all 8 B frags (+ next A frag) issued before the WMMA
// chain so ds-load latency overlaps matrix issue (staggered s_wait_dscnt).
template <int KSTEPS>
__device__ __forceinline__ void mma_rowtile(const __bf16* A, int astride,
                                            const __bf16* BT, int bstride,
                                            int row0, int lane, v8f acc[8]) {
    v16bf a = load_a_frag(A, astride, row0, 0, lane);
    #pragma unroll
    for (int ks = 0; ks < KSTEPS; ++ks) {
        v16bf bfrag[8];
        #pragma unroll
        for (int nt = 0; nt < 8; ++nt)
            bfrag[nt] = load_b_frag(BT, bstride, nt * 16, ks * 32, lane);
        v16bf an = a;
        if (ks + 1 < KSTEPS)
            an = load_a_frag(A, astride, row0, (ks + 1) * 32, lane);
        #pragma unroll
        for (int nt = 0; nt < 8; ++nt)
            acc[nt] = __builtin_amdgcn_wmma_f32_16x16x32_bf16(
                false, a, false, bfrag[nt], (short)0, acc[nt], false, false);
        a = an;
    }
}

extern "C" __global__ __launch_bounds__(256, 1)
void fused_edge_ffn_kernel(
    const float* __restrict__ edge_features,   // [N,32,256]
    const int*   __restrict__ neighbor_mask,   // [N,32]
    const float* __restrict__ W_lin, const float* __restrict__ b_lin,
    const float* __restrict__ g_edge, const float* __restrict__ b_edge,
    const float* __restrict__ W1, const float* __restrict__ b1,
    const float* __restrict__ W2, const float* __restrict__ b2,
    const float* __restrict__ g_node, const float* __restrict__ b_node,
    float* __restrict__ node_out,              // [N,128]
    float* __restrict__ edge_out)              // [N,32,128]
{
    extern __shared__ char smem[];
    __bf16* sWlinT = (__bf16*)(smem + OFF_WLINT);
    __bf16* sW1T   = (__bf16*)(smem + OFF_W1T);
    __bf16* sW2T   = (__bf16*)(smem + OFF_W2T);
    __bf16* sA     = (__bf16*)(smem + OFF_SA);
    float*  sHid   = (float*)(smem + OFF_SHID);
    __bf16* sX     = (__bf16*)(smem + OFF_SX);
    float*  sNode  = (float*)(smem + OFF_SNODE);
    float*  sBias  = (float*)(smem + OFF_BIAS);
    float*  sb_lin  = sBias + 0 * H_DIM;
    float*  sg_edge = sBias + 1 * H_DIM;
    float*  sb_edge = sBias + 2 * H_DIM;
    float*  sb1     = sBias + 3 * H_DIM;
    float*  sb2     = sBias + 4 * H_DIM;
    float*  sg_node = sBias + 5 * H_DIM;
    float*  sb_node = sBias + 6 * H_DIM;
    float*  sNmu   = (float*)(smem + OFF_NSTAT);
    float*  sNrs   = sNmu + NODES_PER_BLK;
    int*    sMask  = (int*)(smem + OFF_MASK);

    const int tid  = threadIdx.x;
    const int lane = tid & 31;
    const int wave = tid >> 5;
    const int row0 = wave * 16;                         // this wave's M-tile
    const long tok0 = (long)blockIdx.x * TOK_PER_BLK;   // first edge-token

    // ---- stage weights (transposed, bf16) with strided scalar loads + packed cvt ----
    #pragma unroll 4
    for (int i = tid; i < (E_DIM * H_DIM) / 4; i += 256) {  // W_lin [e][h]->[h][e]
        int h = i & 127, e4 = (i >> 7) << 2;
        v4f w;
        w.x = W_lin[(e4 + 0) * H_DIM + h];
        w.y = W_lin[(e4 + 1) * H_DIM + h];
        w.z = W_lin[(e4 + 2) * H_DIM + h];
        w.w = W_lin[(e4 + 3) * H_DIM + h];
        *(v4bf*)&sWlinT[h * E_DIM + e4] = __builtin_convertvector(w, v4bf);
    }
    #pragma unroll 4
    for (int i = tid; i < (H_DIM * H_DIM) / 4; i += 256) {  // W1/W2 [h][f]->[f][h]
        int f = i & 127, h4 = (i >> 7) << 2;
        v4f w1, w2;
        w1.x = W1[(h4 + 0) * H_DIM + f];  w2.x = W2[(h4 + 0) * H_DIM + f];
        w1.y = W1[(h4 + 1) * H_DIM + f];  w2.y = W2[(h4 + 1) * H_DIM + f];
        w1.z = W1[(h4 + 2) * H_DIM + f];  w2.z = W2[(h4 + 2) * H_DIM + f];
        w1.w = W1[(h4 + 3) * H_DIM + f];  w2.w = W2[(h4 + 3) * H_DIM + f];
        *(v4bf*)&sW1T[f * H_DIM + h4] = __builtin_convertvector(w1, v4bf);
        *(v4bf*)&sW2T[f * H_DIM + h4] = __builtin_convertvector(w2, v4bf);
    }
    if (tid < H_DIM) {
        sb_lin[tid]  = b_lin[tid];  sg_edge[tid] = g_edge[tid];
        sb_edge[tid] = b_edge[tid]; sb1[tid] = b1[tid]; sb2[tid] = b2[tid];
        sg_node[tid] = g_node[tid]; sb_node[tid] = b_node[tid];
        sMask[tid] = neighbor_mask[(long)blockIdx.x * TOK_PER_BLK + tid];
    }
    // ---- stage A tile: contiguous float4 loads, packed bf16 stores ----
    {
        const float* src = edge_features + tok0 * E_DIM;
        #pragma unroll 4
        for (int i = tid; i < (TOK_PER_BLK * E_DIM) / 4; i += 256) {
            v4f v = *(const v4f*)&src[i * 4];
            *(v4bf*)&sA[i * 4] = __builtin_convertvector(v, v4bf);
        }
    }
    __syncthreads();

    // ---- GEMM1: edge_hidden = A(128x256) * W_lin(256x128), bf16->f32 WMMA ----
    {
        v8f acc[8] = {};
        mma_rowtile<8>(sA, E_DIM, sWlinT, E_DIM, row0, lane, acc);
        const int rb = row0 + ((lane >> 4) << 3);
        #pragma unroll
        for (int nt = 0; nt < 8; ++nt) {
            const int col = nt * 16 + (lane & 15);
            const float bl = sb_lin[col];
            #pragma unroll
            for (int g = 0; g < 8; ++g)
                sHid[(rb + g) * SHID_STRIDE + col] = acc[nt][g] + bl;
        }
    }
    __syncthreads();

    // ---- per-row LayerNorm -> bf16 X ----
    if (tid < TOK_PER_BLK) {
        const float* r = &sHid[tid * SHID_STRIDE];
        float mu = 0.f;
        #pragma unroll 4
        for (int c = 0; c < H_DIM; ++c) mu += r[c];
        mu *= (1.f / H_DIM);
        float var = 0.f;
        #pragma unroll 4
        for (int c = 0; c < H_DIM; ++c) { float d = r[c] - mu; var += d * d; }
        var *= (1.f / H_DIM);
        const float rs = rsqrtf(var + EPSV);
        #pragma unroll 4
        for (int c = 0; c < H_DIM; ++c)
            sX[tid * SX_STRIDE + c] =
                (__bf16)((r[c] - mu) * rs * sg_edge[c] + sb_edge[c]);
    }
    __syncthreads();

    // ---- GEMM2: X * W1, + b1, gelu(tanh approx) -> overwrite own rows of X ----
    {
        v8f acc[8] = {};
        mma_rowtile<4>(sX, SX_STRIDE, sW1T, H_DIM, row0, lane, acc);
        const int rb = row0 + ((lane >> 4) << 3);
        #pragma unroll
        for (int nt = 0; nt < 8; ++nt) {
            const int col = nt * 16 + (lane & 15);
            const float b = sb1[col];
            #pragma unroll
            for (int g = 0; g < 8; ++g) {
                float x = acc[nt][g] + b;
                float t = 0.7978845608028654f * (x + 0.044715f * x * x * x);
                sX[(rb + g) * SX_STRIDE + col] = (__bf16)(0.5f * x * (1.f + tanhf(t)));
            }
        }
    }
    // wave-local RAW on its own 16 rows only -> no barrier needed

    // ---- GEMM3: gelu * W2, + b2, residual -> edge_out tile in sHid ----
    {
        v8f acc[8] = {};
        mma_rowtile<4>(sX, SX_STRIDE, sW2T, H_DIM, row0, lane, acc);
        const int rb = row0 + ((lane >> 4) << 3);
        #pragma unroll
        for (int nt = 0; nt < 8; ++nt) {
            const int col = nt * 16 + (lane & 15);
            const float b = sb2[col];
            #pragma unroll
            for (int g = 0; g < 8; ++g) {
                float* p = &sHid[(rb + g) * SHID_STRIDE + col];
                *p = *p + acc[nt][g] + b;          // edge_hidden + FFN
            }
        }
    }
    __syncthreads();

    // ---- coalesced edge_output write (b128 stores) ----
    {
        const long ebase = tok0 * H_DIM;
        #pragma unroll 4
        for (int i = tid; i < TOK_PER_BLK * H_DIM / 4; i += 256) {
            int r = i >> 5, c4 = (i & 31) << 2;
            *(v4f*)&edge_out[ebase + (long)r * H_DIM + c4] =
                *(const v4f*)&sHid[r * SHID_STRIDE + c4];
        }
    }

    // ---- masked neighbor aggregation: 4 nodes x 128 cols ----
    for (int i = tid; i < NODES_PER_BLK * H_DIM; i += 256) {
        int j = i >> 7, c = i & 127;
        float s = 0.f;
        #pragma unroll
        for (int k = 0; k < K_NB; ++k)
            if (sMask[j * K_NB + k])
                s += sHid[(j * K_NB + k) * SHID_STRIDE + c];
        sNode[j * H_DIM + c] = s;
    }
    __syncthreads();

    // ---- node LayerNorm ----
    if (tid < NODES_PER_BLK) {
        const float* r = &sNode[tid * H_DIM];
        float mu = 0.f;
        for (int c = 0; c < H_DIM; ++c) mu += r[c];
        mu *= (1.f / H_DIM);
        float var = 0.f;
        for (int c = 0; c < H_DIM; ++c) { float d = r[c] - mu; var += d * d; }
        var *= (1.f / H_DIM);
        sNmu[tid] = mu;
        sNrs[tid] = rsqrtf(var + EPSV);
    }
    __syncthreads();
    for (int i = tid; i < NODES_PER_BLK * H_DIM; i += 256) {
        int j = i >> 7, c = i & 127;
        float v = (sNode[j * H_DIM + c] - sNmu[j]) * sNrs[j] * sg_node[c] + sb_node[c];
        node_out[((long)blockIdx.x * NODES_PER_BLK + j) * H_DIM + c] = v;
    }
}

extern "C" void kernel_launch(void* const* d_in, const int* in_sizes, int n_in,
                              void* d_out, int out_size, void* d_ws, size_t ws_size,
                              hipStream_t stream) {
    const float* edge_features = (const float*)d_in[0];
    const int*   neighbor_mask = (const int*)d_in[1];
    const float* W_lin  = (const float*)d_in[2];
    const float* b_lin  = (const float*)d_in[3];
    const float* g_edge = (const float*)d_in[4];
    const float* b_edge = (const float*)d_in[5];
    const float* W1     = (const float*)d_in[6];
    const float* b1     = (const float*)d_in[7];
    const float* W2     = (const float*)d_in[8];
    const float* b2     = (const float*)d_in[9];
    const float* g_node = (const float*)d_in[10];
    const float* b_node = (const float*)d_in[11];

    const int N = in_sizes[1] / K_NB;          // 20000 nodes
    float* node_out = (float*)d_out;           // [N,128] first in tuple order
    float* edge_out = (float*)d_out + (size_t)N * H_DIM;  // [N,32,128]

    hipFuncSetAttribute((const void*)fused_edge_ffn_kernel,
                        hipFuncAttributeMaxDynamicSharedMemorySize, SMEM_BYTES);

    dim3 grid(N / NODES_PER_BLK), block(256);
    fused_edge_ffn_kernel<<<grid, block, SMEM_BYTES, stream>>>(
        edge_features, neighbor_mask, W_lin, b_lin, g_edge, b_edge,
        W1, b1, W2, b2, g_node, b_node, node_out, edge_out);
}
